// WindowZigZagAttention_36352603193768
// MI455X (gfx1250) — compile-verified, matrix operand order checked by reading, not run on hardware
//
#include <hip/hip_runtime.h>
#include <hip/hip_bf16.h>
#include <stdint.h>

// ---------- types ----------
typedef __attribute__((ext_vector_type(8)))  __bf16 v8bf;
typedef __attribute__((ext_vector_type(16))) __bf16 v16bf;
typedef __attribute__((ext_vector_type(8)))  float  v8f;

__device__ __forceinline__ uint16_t f2bf(float f) {
  uint32_t u = __float_as_uint(f);
  uint32_t r = u + 0x7FFFu + ((u >> 16) & 1u);   // round-to-nearest-even
  return (uint16_t)(r >> 16);
}

// Load one 16x32 bf16 WMMA A/B fragment from a row-major [rows][ld] bf16 matrix.
// Lane L holds row (row0 + (L&15)); elements 0..7 = K[kb..kb+7], 8..15 = K[16+kb..],
// kb = (L>>4)*8  (CDNA5 ISA 16-bit A-matrix layout; B uses the same
// per-lane pattern indexed by its N column, i.e. load B^T rows).
__device__ __forceinline__ v16bf frag_ld(const uint16_t* base, int row0, int k0, int ld) {
  const int lane = threadIdx.x & 31;
  const int kb   = (lane >> 4) << 3;
  const uint16_t* p = base + (size_t)(row0 + (lane & 15)) * ld + k0 + kb;
  v8bf lo = *(const v8bf*)(p);
  v8bf hi = *(const v8bf*)(p + 16);
  return __builtin_shufflevector(lo, hi, 0,1,2,3,4,5,6,7,8,9,10,11,12,13,14,15);
}

__device__ __forceinline__ v8f wmma_bf16(v16bf a, v16bf b, v8f c) {
  return __builtin_amdgcn_wmma_f32_16x16x32_bf16(false, a, false, b, (short)0, c, false, false);
}

// Async memory->LDS copy, 16B per lane, tracked by ASYNCcnt (CDNA5 ISA).
// lds generic pointer's low 32 bits are the wave-relative LDS byte address
// (flat->LDS aperture mapping truncates to addr[31:0], CDNA5 ISA).
__device__ __forceinline__ void async_copy_b128(void* lds, const void* gaddr) {
  uint32_t l = (uint32_t)(uintptr_t)lds;
  asm volatile("global_load_async_to_lds_b128 %0, %1, off"
               :: "v"(l), "v"(gaddr) : "memory");
}
__device__ __forceinline__ void wait_async0() {
  asm volatile("s_wait_asynccnt 0x0" ::: "memory");
}

// ---------- prep kernels ----------
__global__ __launch_bounds__(256) void k_f2bf_arr(const float* __restrict__ src,
                                                  uint16_t* __restrict__ dst, int n) {
  int i = blockIdx.x * 256 + threadIdx.x;
  if (i < n) dst[i] = f2bf(src[i]);
}

// dst[n*K + k] = bf16(src[k*N + n])   (W^T so B-fragments read contiguous K)
__global__ __launch_bounds__(256) void k_wT(const float* __restrict__ src,
                                            uint16_t* __restrict__ dst, int K, int N) {
  int i = blockIdx.x * 256 + threadIdx.x;
  if (i < K * N) {
    int n = i / K, k = i - n * K;
    dst[i] = f2bf(src[(size_t)k * N + n]);
  }
}

__global__ void k_zigzag(int* __restrict__ coords) {  // coords[0..255]=r, [256..511]=c
  if (threadIdx.x == 0 && blockIdx.x == 0) {
    int r = 0, c = 0;
    for (int i = 0; i < 256; ++i) {
      coords[i] = r; coords[256 + i] = c;
      if (((r + c) & 1) == 0) {
        if (c == 15) r += 1;
        else if (r == 0) c += 1;
        else { r -= 1; c += 1; }
      } else if (r == 15) { c += 1; }
      else if (c == 0)    { r += 1; }
      else { r += 1; c -= 1; }
    }
  }
}

// biasM[h][n][m] = bias_table[relidx(n,m)][h]
__global__ __launch_bounds__(256) void k_bias(const int* __restrict__ coords,
                                              const float* __restrict__ table,
                                              float* __restrict__ biasM) {
  int h = blockIdx.x >> 8, n = blockIdx.x & 255, m = threadIdx.x;
  int hh = coords[n]       - coords[m]       + 15;
  int ww = coords[256 + n] - coords[256 + m] + 15;
  biasM[((size_t)(h * 256 + n)) * 256 + m] = table[(hh * 31 + ww) * 16 + h];
}

// ---------- stage 1: QKV GEMM (65536x512 @ 512x1536), scatter q/k/v ----------
__global__ __launch_bounds__(256) void k_gemm_qkv(const uint16_t* __restrict__ Xb,
                                                  const uint16_t* __restrict__ WT,
                                                  const float* __restrict__ bias,
                                                  uint16_t* __restrict__ Qb,
                                                  uint16_t* __restrict__ Kb,
                                                  uint16_t* __restrict__ Vt) {
  const int wave = threadIdx.x >> 5, lane = threadIdx.x & 31;
  const int row0 = blockIdx.x * 128 + wave * 16;
  const int col0 = blockIdx.y * 64;
  v8f acc[4] = {};
  for (int k = 0; k < 512; k += 32) {
    v16bf a = frag_ld(Xb, row0, k, 512);
#pragma unroll
    for (int t = 0; t < 4; ++t) {
      v16bf b = frag_ld(WT, col0 + 16 * t, k, 512);
      acc[t] = wmma_bf16(a, b, acc[t]);
    }
  }
  const float scale = 0.17677669529663687f;  // 32^-0.5 folded into Q
  const int rbase = row0 + ((lane >> 4) << 3);
#pragma unroll
  for (int t = 0; t < 4; ++t) {
    int col = col0 + 16 * t + (lane & 15);
    int which = col >> 9, h = (col >> 5) & 15, d = col & 31;
    float bv = bias[col];
#pragma unroll
    for (int i = 0; i < 8; ++i) {
      int r = rbase + i, b = r >> 8, n = r & 255;
      float v = acc[t][i] + bv;
      size_t bh = (size_t)(b * 16 + h);
      if (which == 0)      Qb[(bh * 256 + n) * 32 + d] = f2bf(v * scale);
      else if (which == 1) Kb[(bh * 256 + n) * 32 + d] = f2bf(v);
      else                 Vt[(bh * 32 + d) * 256 + n] = f2bf(v);  // V transposed
    }
  }
}

// ---------- stage 2: attention, one block per (b,h) ----------
#define SROW 264             // padded f32 row stride (264%64==8 -> 4-way LDS conflicts)
#define TILE_BYTES 16384     // 256x32 bf16 (K) == 32x256 bf16 (V^T)
#define S_OFF (2 * TILE_BYTES)
#define ATTN_LDS (S_OFF + 256 * SROW * 4)   // 303104 B <= 320 KB/WGP
__global__ __launch_bounds__(256) void k_attn(const uint16_t* __restrict__ Qb,
                                              const uint16_t* __restrict__ Kb,
                                              const uint16_t* __restrict__ Vt,
                                              const float* __restrict__ biasM,
                                              uint16_t* __restrict__ Hb) {
  extern __shared__ char smem[];
  uint16_t* Klds = (uint16_t*)smem;                 // [256][32] bf16
  uint16_t* Vlds = (uint16_t*)(smem + TILE_BYTES);  // [32][256] bf16
  float*    S    = (float*)(smem + S_OFF);          // [256][SROW] f32 scores
  uint16_t* P    = (uint16_t*)(smem + S_OFF);       // bf16 probs interleaved over S
  const int bh = blockIdx.x, h = bh & 15, b = bh >> 4;
  const int wave = threadIdx.x >> 5, lane = threadIdx.x & 31;
  const uint16_t* Qw = Qb + (size_t)bh * 256 * 32;
  const uint16_t* Kw = Kb + (size_t)bh * 256 * 32;
  const uint16_t* Vw = Vt + (size_t)bh * 32 * 256;

  // ---- async-stage K and V^T tiles into LDS (memory->LDS, no VGPR round trip) ----
#pragma unroll
  for (int i = 0; i < 4; ++i) {
    uint32_t o = (threadIdx.x + 256 * i) * 16;      // 4 KB per sweep x 4 = 16 KB
    async_copy_b128((char*)Klds + o, (const char*)Kw + o);
    async_copy_b128((char*)Vlds + o, (const char*)Vw + o);
  }
  wait_async0();
  __syncthreads();

  // ---- S = Q * K^T  (hd=32 == exactly one 16x16x32 WMMA per tile) ----
#pragma unroll
  for (int tr = 0; tr < 2; ++tr) {
    const int n0 = (wave * 2 + tr) * 16;
    v16bf a = frag_ld(Qw, n0, 0, 32);
    for (int tm = 0; tm < 16; ++tm) {
      v16bf bk = frag_ld(Klds, tm * 16, 0, 32);
      v8f c = {};
      c = wmma_bf16(a, bk, c);
      int m  = tm * 16 + (lane & 15);
      int rb = n0 + ((lane >> 4) << 3);
#pragma unroll
      for (int i = 0; i < 8; ++i) S[(rb + i) * SROW + m] = c[i];
    }
  }
  __syncthreads();

  // ---- softmax row-per-thread (+bias), write bf16 P in place over consumed S ----
  {
    const int n = threadIdx.x;
    float* Srow = S + (size_t)n * SROW;
    const float* brow = biasM + ((size_t)h * 256 + n) * 256;
    float mx = -3.4e38f;
    for (int m = 0; m < 256; ++m) { float v = Srow[m] + brow[m]; Srow[m] = v; mx = fmaxf(mx, v); }
    float sum = 0.f;
    for (int m = 0; m < 256; ++m) { float e = __expf(Srow[m] - mx); Srow[m] = e; sum += e; }
    float inv = 1.0f / sum;
    uint16_t* Prow = P + (size_t)n * (SROW * 2);
    for (int m = 0; m < 256; ++m) Prow[m] = f2bf(Srow[m] * inv);  // writes idx m/2 < reads idx m
  }
  __syncthreads();

  // ---- O = P * V  (K=256 -> 8 WMMA steps), store [b, n, h*32+d] ----
#pragma unroll
  for (int tr = 0; tr < 2; ++tr) {
    const int n0 = (wave * 2 + tr) * 16;
#pragma unroll
    for (int td = 0; td < 2; ++td) {
      v8f acc = {};
      for (int k = 0; k < 256; k += 32) {
        v16bf a = frag_ld(P, n0, k, SROW * 2);
        v16bf v = frag_ld(Vlds, td * 16, k, 256);
        acc = wmma_bf16(a, v, acc);
      }
      int col = h * 32 + td * 16 + (lane & 15);
      int nb  = n0 + ((lane >> 4) << 3);
#pragma unroll
      for (int i = 0; i < 8; ++i)
        Hb[((size_t)(b * 256 + nb + i)) * 512 + col] = f2bf(acc[i]);
    }
  }
}

// ---------- stage 3: proj GEMM (65536x512 @ 512x512) + bias -> f32 out ----------
__global__ __launch_bounds__(256) void k_gemm_proj(const uint16_t* __restrict__ Hb,
                                                   const uint16_t* __restrict__ WT,
                                                   const float* __restrict__ bias,
                                                   float* __restrict__ out) {
  const int wave = threadIdx.x >> 5, lane = threadIdx.x & 31;
  const int row0 = blockIdx.x * 128 + wave * 16;
  const int col0 = blockIdx.y * 64;
  v8f acc[4] = {};
  for (int k = 0; k < 512; k += 32) {
    v16bf a = frag_ld(Hb, row0, k, 512);
#pragma unroll
    for (int t = 0; t < 4; ++t) {
      v16bf b = frag_ld(WT, col0 + 16 * t, k, 512);
      acc[t] = wmma_bf16(a, b, acc[t]);
    }
  }
  const int rbase = row0 + ((lane >> 4) << 3);
#pragma unroll
  for (int t = 0; t < 4; ++t) {
    int col = col0 + 16 * t + (lane & 15);
    float bv = bias[col];
#pragma unroll
    for (int i = 0; i < 8; ++i)
      out[(size_t)(rbase + i) * 512 + col] = acc[t][i] + bv;
  }
}

// ---------- host ----------
extern "C" void kernel_launch(void* const* d_in, const int* in_sizes, int n_in,
                              void* d_out, int out_size, void* d_ws, size_t ws_size,
                              hipStream_t stream) {
  const float* x      = (const float*)d_in[0];   // [65536][512]
  const float* qkv_w  = (const float*)d_in[1];   // [512][1536]
  const float* qkv_b  = (const float*)d_in[2];   // [1536]
  const float* proj_w = (const float*)d_in[3];   // [512][512]
  const float* proj_b = (const float*)d_in[4];   // [512]
  const float* btab   = (const float*)d_in[5];   // [961][16]
  float* out = (float*)d_out;

  uint8_t* ws = (uint8_t*)d_ws;
  size_t off = 0;
  auto carve = [&](size_t bytes) { uint8_t* p = ws + off; off = (off + bytes + 255) & ~(size_t)255; return p; };
  uint16_t* Xb     = (uint16_t*)carve((size_t)65536 * 512 * 2);   // 67 MB
  uint16_t* WqkvT  = (uint16_t*)carve((size_t)1536 * 512 * 2);
  uint16_t* WprojT = (uint16_t*)carve((size_t)512 * 512 * 2);
  uint16_t* Qb     = (uint16_t*)carve((size_t)65536 * 512 * 2);
  uint16_t* Kb     = (uint16_t*)carve((size_t)65536 * 512 * 2);
  uint16_t* Vt     = (uint16_t*)carve((size_t)65536 * 512 * 2);
  uint16_t* Hb     = (uint16_t*)carve((size_t)65536 * 512 * 2);
  float*    biasM  = (float*)carve((size_t)16 * 256 * 256 * 4);   // 4 MB
  int*      coords = (int*)carve(512 * 4);
  (void)ws_size; (void)in_sizes; (void)n_in; (void)out_size;

  // prep
  k_f2bf_arr<<<(65536 * 512) / 256, 256, 0, stream>>>(x, Xb, 65536 * 512);
  k_wT<<<(1536 * 512) / 256, 256, 0, stream>>>(qkv_w, WqkvT, 512, 1536);
  k_wT<<<(512 * 512) / 256, 256, 0, stream>>>(proj_w, WprojT, 512, 512);
  k_zigzag<<<1, 32, 0, stream>>>(coords);
  k_bias<<<16 * 256, 256, 0, stream>>>(coords, btab, biasM);

  // stage 1: QKV
  k_gemm_qkv<<<dim3(65536 / 128, 1536 / 64), 256, 0, stream>>>(Xb, WqkvT, qkv_b, Qb, Kb, Vt);

  // stage 2: attention (303104 B dynamic LDS, fits 320 KB/WGP)
  k_attn<<<256 * 16, 256, ATTN_LDS, stream>>>(Qb, Kb, Vt, biasM, Hb);

  // stage 3: proj
  k_gemm_proj<<<dim3(65536 / 128, 512 / 64), 256, 0, stream>>>(Hb, WprojT, proj_b, out);
}
